// EngramCache_10453950398504
// MI455X (gfx1250) — matrix operand order repaired and checked by reading, not compiled, not run on hardware
//
#include <hip/hip_runtime.h>

typedef __attribute__((ext_vector_type(16))) _Float16 v16h;
typedef __attribute__((ext_vector_type(8)))  float    v8f;

#define TSZ 100000
#define DK  2048          // K = H*2*E
#define DD  2048          // hidden/out dim
#define TT  4096
#define VP  2056          // LDS v-stash pitch (halves), padded for bank spread

union FragU { uint4 q[2]; v16h v; };
union H8U  { uint4 q; _Float16 h[8]; };
union H4U  { uint2 d; _Float16 h[4]; };
union HSU  { _Float16 h; unsigned short s; };

// ------------------------------------------------------------------
// Kernel 1: repack Wv [N=2048, K=2048] fp32 -> f16 B-fragment-major.
// fragment fb = kt*128 + ntAbs ; Bf[(fb*32+lane)*16 + j] holds
//   n = ntAbs*16 + (lane&15), k = kt*32 + (lane<16 ? j : 16+j)
// (B 32x16 as packed here: lane = column, 16 consecutive K per lane.)
// ------------------------------------------------------------------
__global__ void repack_wv(const float* __restrict__ Wv, _Float16* __restrict__ Bf) {
    int tid  = blockIdx.x * blockDim.x + threadIdx.x;   // 8192 frags * 32 lanes
    int lane = tid & 31;
    int f    = tid >> 5;
    int kt   = f >> 7;
    int ntA  = f & 127;
    int n    = ntA * 16 + (lane & 15);
    int kb   = kt * 32 + ((lane & 16) ? 16 : 0);
    const float* src = Wv + (size_t)n * DK + kb;
    _Float16* dst = Bf + (size_t)(f * 32 + lane) * 16;
    alignas(16) _Float16 tmp[16];
#pragma unroll
    for (int j = 0; j < 16; ++j) tmp[j] = (_Float16)src[j];
    *(uint4*)(dst)     = *(const uint4*)(&tmp[0]);
    *(uint4*)(dst + 8) = *(const uint4*)(&tmp[8]);
}

// ------------------------------------------------------------------
// Kernel 2: hash + table gather -> e as f16 A-fragment-major in ws.
// fragment fa = mt*64 + kt ; A 16x32 as packed here:
//   lane<16: row=lane, halves j0..15 = K kt*32+0..15
//   lane>=16: K+16..31
// ------------------------------------------------------------------
__global__ void gather_pack(const float* __restrict__ t2,
                            const float* __restrict__ t3,
                            const int* __restrict__ ids,
                            _Float16* __restrict__ Af) {
    __shared__ int I2[16], I3[16];
    const int tid = threadIdx.x;
    const int mt  = blockIdx.x;          // 0..1023 (16 rows each)
    const int m0  = mt * 16;
    if (tid < 16) {
        int m   = m0 + tid;
        int t   = m & (TT - 1);
        int id0 = min(max(ids[m], 0), 50256);
        int p1  = (t >= 1) ? min(max(ids[m - 1], 0), 50256) : 0;
        int p2  = (t >= 2) ? min(max(ids[m - 2], 0), 50256) : 0;
        int h2  = (id0 * 20011) ^ (p1 * 30011);
        int h3  = h2 ^ (p2 * 40009);
        I2[tid] = h2 % TSZ;
        I3[tid] = h3 % TSZ;
    }
    __syncthreads();
    for (int it = 0; it < 32; ++it) {
        int g = it * 256 + tid;          // 8192 float4 chunks (16 rows x 512)
        int r = g >> 9;
        int k = (g & 511) << 2;
        const float* src;
        if (k < 1024) {
            src = t2 + ((size_t)(k >> 8) * TSZ + I2[r]) * 256 + (k & 255);
        } else {
            int kk = k - 1024;
            src = t3 + ((size_t)(kk >> 8) * TSZ + I3[r]) * 256 + (kk & 255);
        }
        float4 fv = *(const float4*)src;
        H4U hv;
        hv.h[0] = (_Float16)fv.x; hv.h[1] = (_Float16)fv.y;
        hv.h[2] = (_Float16)fv.z; hv.h[3] = (_Float16)fv.w;
        int kt   = k >> 5;
        int k4   = k & 31;
        int run  = k4 >> 3;
        int lane = (r & 15) + ((run & 2) << 3);
        int j    = ((run & 1) << 3) + (k4 & 7);
        size_t dst = ((size_t)(mt * 64 + kt) * 32 + lane) * 16 + j;
        *(uint2*)&Af[dst] = hv.d;
    }
}

// ------------------------------------------------------------------
// Kernel 3: WMMA GEMM (64 rows x 2048 cols per workgroup, 4 N-passes,
// double-buffered fragment streams) + fused RMS-gate epilogue.
// ------------------------------------------------------------------
__launch_bounds__(256, 1)
__global__ void gemm_gate(const float* __restrict__ hidden,
                          const _Float16* __restrict__ BfH,
                          const _Float16* __restrict__ AfH,
                          const float* __restrict__ gwh,
                          const float* __restrict__ gwv,
                          float* __restrict__ out) {
    __shared__ __align__(16) unsigned short Vh[64 * VP];   // ~257 KB f16 v-stash
    __shared__ float Shh[64], Svv[64], Shv[64], Gate[64];

    const int tid  = threadIdx.x;
    const int lane = tid & 31;
    const int w    = tid >> 5;          // wave 0..7
    const int mb   = blockIdx.x;        // 0..255
    const int m0   = mb * 64;
    const int mt0  = mb * 4;            // first 16-row M-tile

    if (tid < 64) { Shh[tid] = 0.f; Svv[tid] = 0.f; Shv[tid] = 0.f; }

    const uint4* Ab = (const uint4*)AfH + (size_t)lane * 2;
    const uint4* Bb = (const uint4*)BfH + (size_t)lane * 2;
    const int rowOff = (lane & 16) ? 8 : 0;
    const int nlo    = lane & 15;

#pragma unroll 1
    for (int p = 0; p < 4; ++p) {
        const int ntA0 = p * 32 + w * 4;       // wave's 4 N-tiles this pass
        v8f acc[16];
        {
            v8f z = {0.f, 0.f, 0.f, 0.f, 0.f, 0.f, 0.f, 0.f};
#pragma unroll
            for (int i = 0; i < 16; ++i) acc[i] = z;
        }

        uint4 AX[8], BX[8], AY[8], BY[8];
        // prologue: kt=0 -> X buffers
#pragma unroll
        for (int mi = 0; mi < 4; ++mi) {
            size_t f = (size_t)((mt0 + mi) * 64) * 64;
            AX[2*mi] = Ab[f]; AX[2*mi+1] = Ab[f + 1];
        }
#pragma unroll
        for (int nt = 0; nt < 4; ++nt) {
            size_t f = (size_t)(ntA0 + nt) * 64;
            BX[2*nt] = Bb[f]; BX[2*nt+1] = Bb[f + 1];
        }

#pragma unroll 1
        for (int kt = 0; kt < 64; kt += 2) {
            // stream kt+1 into Y while computing kt from X
            int k1 = kt + 1;
#pragma unroll
            for (int mi = 0; mi < 4; ++mi) {
                size_t f = (size_t)((mt0 + mi) * 64 + k1) * 64;
                AY[2*mi] = Ab[f]; AY[2*mi+1] = Ab[f + 1];
            }
#pragma unroll
            for (int nt = 0; nt < 4; ++nt) {
                size_t f = (size_t)(k1 * 128 + ntA0 + nt) * 64;
                BY[2*nt] = Bb[f]; BY[2*nt+1] = Bb[f + 1];
            }
#pragma unroll
            for (int mi = 0; mi < 4; ++mi) {
                FragU a; a.q[0] = AX[2*mi]; a.q[1] = AX[2*mi+1];
#pragma unroll
                for (int nt = 0; nt < 4; ++nt) {
                    FragU b; b.q[0] = BX[2*nt]; b.q[1] = BX[2*nt+1];
                    acc[mi*4+nt] = __builtin_amdgcn_wmma_f32_16x16x32_f16(
                        false, a.v, false, b.v, (short)0, acc[mi*4+nt], false, false);
                }
            }
            // stream kt+2 into X while computing kt+1 from Y
            int k2 = (kt + 2 < 64) ? kt + 2 : 0;   // tail prefetch clamped in-bounds
#pragma unroll
            for (int mi = 0; mi < 4; ++mi) {
                size_t f = (size_t)((mt0 + mi) * 64 + k2) * 64;
                AX[2*mi] = Ab[f]; AX[2*mi+1] = Ab[f + 1];
            }
#pragma unroll
            for (int nt = 0; nt < 4; ++nt) {
                size_t f = (size_t)(k2 * 128 + ntA0 + nt) * 64;
                BX[2*nt] = Bb[f]; BX[2*nt+1] = Bb[f + 1];
            }
#pragma unroll
            for (int mi = 0; mi < 4; ++mi) {
                FragU a; a.q[0] = AY[2*mi]; a.q[1] = AY[2*mi+1];
#pragma unroll
                for (int nt = 0; nt < 4; ++nt) {
                    FragU b; b.q[0] = BY[2*nt]; b.q[1] = BY[2*nt+1];
                    acc[mi*4+nt] = __builtin_amdgcn_wmma_f32_16x16x32_f16(
                        false, a.v, false, b.v, (short)0, acc[mi*4+nt], false, false);
                }
            }
        }

        // stash this pass's 64x512 result slab to LDS as f16
#pragma unroll
        for (int mi = 0; mi < 4; ++mi)
#pragma unroll
            for (int nt = 0; nt < 4; ++nt) {
                int col = p * 512 + w * 64 + nt * 16 + nlo;
#pragma unroll
                for (int r = 0; r < 8; ++r) {
                    int row = mi * 16 + r + rowOff;
                    HSU u; u.h = (_Float16)acc[mi*4+nt][r];
                    Vh[row * VP + col] = u.s;
                }
            }
    }
    __syncthreads();

    // ---- per-row RMS stats: 4 threads per row, 512 cols each ----
    {
        int row = tid >> 2;
        int c   = tid & 3;
        float shh = 0.f, svv = 0.f, shv = 0.f;
        const float* hrow = hidden + (size_t)(m0 + row) * DD;
        for (int i = 0; i < 64; ++i) {
            int cb = c * 512 + i * 8;
            H8U vv; vv.q = *(const uint4*)&Vh[row * VP + cb];
            float4 h0 = *(const float4*)(hrow + cb);
            float4 h1 = *(const float4*)(hrow + cb + 4);
            float4 g0 = *(const float4*)(gwh + cb);
            float4 g1 = *(const float4*)(gwh + cb + 4);
            float4 q0 = *(const float4*)(gwv + cb);
            float4 q1 = *(const float4*)(gwv + cb + 4);
            float hs[8] = {h0.x,h0.y,h0.z,h0.w,h1.x,h1.y,h1.z,h1.w};
            float gs[8] = {g0.x,g0.y,g0.z,g0.w,g1.x,g1.y,g1.z,g1.w};
            float qs[8] = {q0.x,q0.y,q0.z,q0.w,q1.x,q1.y,q1.z,q1.w};
#pragma unroll
            for (int j = 0; j < 8; ++j) {
                float v = (float)vv.h[j];
                float h = hs[j];
                shh += h * h;
                svv += v * v;
                shv += h * v * gs[j] * qs[j];
            }
        }
        atomicAdd(&Shh[row], shh);
        atomicAdd(&Svv[row], svv);
        atomicAdd(&Shv[row], shv);
    }
    __syncthreads();

    if (tid < 64) {
        const float eps = 1.1920929e-7f;                     // finfo(f32).eps
        float ih = rsqrtf(Shh[tid] * (1.0f / 2048.0f) + eps);
        float iv = rsqrtf(Svv[tid] * (1.0f / 2048.0f) + eps);
        float x  = Shv[tid] * ih * iv * 0.022097086912079608f; // /sqrt(2048)
        float ax = fabsf(x);
        float sg = (x > 0.f) ? 1.f : ((x < 0.f) ? -1.f : 0.f);
        float y  = sqrtf(fmaxf(ax, 1e-6f)) * sg;
        Gate[tid] = 1.0f / (1.0f + __expf(-y));
    }
    __syncthreads();

    // ---- coalesced gated output: row by row, 1024 floats per iter ----
#pragma unroll 1
    for (int r = 0; r < 64; ++r) {
        float g = Gate[r];
        float* orow = out + (size_t)(m0 + r) * DD;
#pragma unroll
        for (int i = 0; i < 2; ++i) {
            int col = (i * 256 + tid) * 4;
            H4U u; u.d = *(const uint2*)&Vh[r * VP + col];
            float4 o = { g * (float)u.h[0], g * (float)u.h[1],
                         g * (float)u.h[2], g * (float)u.h[3] };
            *(float4*)(orow + col) = o;
        }
    }
}

extern "C" void kernel_launch(void* const* d_in, const int* in_sizes, int n_in,
                              void* d_out, int out_size, void* d_ws, size_t ws_size,
                              hipStream_t stream) {
    const float* hidden = (const float*)d_in[0];
    const float* t2     = (const float*)d_in[1];
    const float* t3     = (const float*)d_in[2];
    const float* Wv     = (const float*)d_in[3];
    const float* gwh    = (const float*)d_in[4];
    const float* gwv    = (const float*)d_in[5];
    const int*   ids    = (const int*)d_in[6];
    float*       out    = (float*)d_out;

    _Float16* Bf = (_Float16*)d_ws;                              // 8 MB
    _Float16* Af = (_Float16*)((char*)d_ws + (8u << 20));        // 64 MB

    repack_wv<<<1024, 256, 0, stream>>>(Wv, Bf);                 // 8192 frags
    gather_pack<<<1024, 256, 0, stream>>>(t2, t3, ids, Af);      // 1024 M-tiles
    gemm_gate<<<256, 256, 0, stream>>>(hidden, Bf, Af, gwh, gwv, out);
}
